// UpModConvBlock_82471962018177
// MI455X (gfx1250) — matrix operand order repaired and checked by reading, not compiled, not run on hardware
//
#include <hip/hip_runtime.h>
#include <hip/hip_bf16.h>
#include <math.h>

// ---- problem constants (B, Cin, Cout, H, W, L) = (16, 512, 512, 32, 32, 512)
#define B_    16
#define CIN   512
#define COUT  512
#define HH    32
#define WW    32
#define LL    512
#define H2    65     // upconv intermediate (65x65)
#define HO    64     // final output (64x64)
#define CINP  514    // padded Cin stride for transpose LDS tile

typedef __attribute__((ext_vector_type(16))) __bf16        v16bf;
typedef __attribute__((ext_vector_type(8)))  float         v8f;
typedef __attribute__((ext_vector_type(4)))  unsigned int  u32x4;

static __device__ __forceinline__ unsigned short f2bf(float f) {
  union { float f; unsigned int u; } c; c.f = f;
  unsigned int u = c.u;
  u += 0x7fffu + ((u >> 16) & 1u);      // round-to-nearest-even
  return (unsigned short)(u >> 16);
}

// ---------------- style[b][ci] = (w[b] . aw[ci]) * lin_scale + ab[ci] ----------------
__global__ void style_kernel(const float* __restrict__ w, const float* __restrict__ aw,
                             const float* __restrict__ ab, float* __restrict__ style,
                             float lin_scale) {
  int idx = blockIdx.x * 256 + threadIdx.x;           // b*CIN + ci
  if (idx >= B_ * CIN) return;
  int b = idx / CIN, ci = idx - b * CIN;
  const float* wr = w + (size_t)b * LL;
  const float* ar = aw + (size_t)ci * LL;
  float acc = 0.f;
  for (int l = 0; l < LL; ++l) acc += wr[l] * ar[l];
  style[idx] = acc * lin_scale + ab[ci];
}

// ---------------- wsq[co][ci] = sum over 9 taps of conv_weight^2 ----------------
__global__ void wsq_kernel(const float* __restrict__ cw, float* __restrict__ wsq) {
  int idx = blockIdx.x * 256 + threadIdx.x;           // co*CIN + ci
  if (idx >= COUT * CIN) return;
  const float* p = cw + (size_t)idx * 9;
  float acc = 0.f;
#pragma unroll
  for (int t = 0; t < 9; ++t) acc += p[t] * p[t];
  wsq[idx] = acc;
}

// ---------------- d[b][co] = rsqrt(scale^2 * sum_ci style^2 * wsq + 1e-8) ----------------
__global__ void demod_kernel(const float* __restrict__ style, const float* __restrict__ wsq,
                             float* __restrict__ dmod, float scale2) {
  int idx = blockIdx.x * 256 + threadIdx.x;           // b*COUT + co
  if (idx >= B_ * COUT) return;
  int b = idx / COUT, co = idx - b * COUT;
  const float* st = style + (size_t)b * CIN;
  const float* wq = wsq + (size_t)co * CIN;
  float acc = 0.f;
  for (int ci = 0; ci < CIN; ++ci) { float s = st[ci]; acc += s * s * wq[ci]; }
  dmod[idx] = rsqrtf(scale2 * acc + 1e-8f);
}

// ---------------- Wb[b][tap][co][ci] = bf16(scale * d * style * cw[co][ci][tap]) ----------------
__global__ void packw_kernel(const float* __restrict__ cw, const float* __restrict__ style,
                             const float* __restrict__ dmod, unsigned short* __restrict__ Wb,
                             float scale) {
  long long idx = (long long)blockIdx.x * 256 + threadIdx.x;   // ((b*9+tap)*COUT+co)*CIN+ci
  const long long total = (long long)B_ * 9 * COUT * CIN;
  if (idx >= total) return;
  int ci  = (int)(idx & (CIN - 1));
  int co  = (int)((idx >> 9) & (COUT - 1));
  int tap = (int)((idx / ((long long)COUT * CIN)) % 9);
  int b   = (int)(idx / ((long long)9 * COUT * CIN));
  float v = scale * dmod[b * COUT + co] * style[b * CIN + ci]
          * cw[((size_t)co * CIN + ci) * 9 + tap];
  Wb[idx] = f2bf(v);
}

// ---------------- transpose input to channels-last bf16: XbT[b][r][c][ci] ----------------
__global__ __launch_bounds__(256)
void transpose_kernel(const float* __restrict__ in, unsigned short* __restrict__ XbT) {
  __shared__ unsigned short tile[WW * CINP];     // [c][ci], padded stride
  const int br = blockIdx.x;                     // b*HH + r
  const int b = br / HH, r = br % HH;
  // read in[b][ci][r][c] (c contiguous -> coalesced), write tile[c][ci]
  for (int i = threadIdx.x; i < CIN * WW; i += 256) {
    int ci = i >> 5, c = i & 31;
    tile[c * CINP + ci] = f2bf(in[(((size_t)b * CIN + ci) * HH + r) * WW + c]);
  }
  __syncthreads();
  // write XbT[b][r][c][ci] (ci contiguous -> coalesced)
  unsigned short* dst = XbT + ((size_t)br * WW) * CIN;
  for (int i = threadIdx.x; i < CIN * WW; i += 256) {
    int c = i >> 9, ci = i & 511;
    dst[i] = tile[c * CINP + ci];
  }
}

// ---------------- upsampling modulated conv via WMMA implicit GEMM ----------------
// Block: 256 threads = 8 waves. Each wave: 32 couts x 16 same-phase pixels of row y.
// gridDim = (5 xtiles [3 even-phase + 2 odd-phase], 65 rows, 16 b * 2 co-blocks)
__global__ __launch_bounds__(256)
void upconv_kernel(const unsigned short* __restrict__ Wb,
                   const unsigned short* __restrict__ XbT,
                   float* __restrict__ o65) {
  // input window, channels-fastest: smem[(s*17 + cc)*CIN + ci]
  __shared__ __align__(16) unsigned short smem[2 * 17 * CIN];   // 34816 B

  const int tid = threadIdx.x;
  const int y   = blockIdx.y;
  const int t   = blockIdx.x;
  const int bz  = blockIdx.z;
  const int b   = bz >> 1;
  const int cb  = bz & 1;

  const int ey = y & 1;
  const int ex = (t < 3) ? 0 : 1;
  const int xt = (t < 3) ? t : (t - 3);
  const int c0 = 16 * xt - 1;                     // leftmost input column needed

  int r0, r1, nrows;
  if (ey == 0) { r0 = y >> 1;       r1 = r0 - 1; nrows = 2; }
  else         { r0 = (y - 1) >> 1; r1 = -1;     nrows = 1; }

  // ---- async stage input window into LDS (16B chunks); zero-fill OOB chunks ----
  // 34 rows of CIN bf16 = 34 * 64 chunks of 16B
  const u32x4 zero4 = {0u, 0u, 0u, 0u};
  for (int i = tid; i < 34 * 64; i += 256) {
    const int row   = i >> 6;                    // s*17 + cc
    const int chunk = i & 63;
    const int s  = row / 17;
    const int cc = row - s * 17;
    const int r  = (s == 0) ? r0 : r1;
    const int c  = c0 + cc;
    unsigned short* lp = &smem[(size_t)i * 8];
    if (s < nrows && r >= 0 && r < HH && c >= 0 && c < WW) {
      const unsigned short* gp =
          XbT + ((((size_t)b * HH + r) * WW + c) * CIN) + chunk * 8;
      unsigned int       loff  = (unsigned int)(size_t)lp;
      unsigned long long gaddr = (unsigned long long)(size_t)gp;
      asm volatile("global_load_async_to_lds_b128 %0, %1, off"
                   :: "v"(loff), "v"(gaddr) : "memory");
    } else {
      *reinterpret_cast<u32x4*>(lp) = zero4;
    }
  }
  asm volatile("s_wait_asynccnt 0x0" ::: "memory");
  __syncthreads();

  const int wv   = tid >> 5;
  const int lane = tid & 31;
  const int m    = lane & 15;                    // A row / B column within tile
  const int hi   = lane >> 4;                    // K-half selector
  const int co0  = cb * 256 + wv * 32;

  v8f acc0 = {0.f, 0.f, 0.f, 0.f, 0.f, 0.f, 0.f, 0.f};
  v8f acc1 = {0.f, 0.f, 0.f, 0.f, 0.f, 0.f, 0.f, 0.f};

  const int nay = ey ? 1 : 2;
  const int nax = ex ? 1 : 2;
  for (int ti = 0; ti < nay; ++ti) {
    const int ay = ey ? 1 : (ti == 0 ? 0 : 2);
    const int s  = ti;                           // LDS row slot
    for (int xj = 0; xj < nax; ++xj) {
      const int ax  = ex ? 1 : (xj == 0 ? 0 : 2);
      const int dx  = (ax == ex) ? 0 : -1;       // input col shift
      const int tap = ay * 3 + ax;
      const int cc  = m + 1 + dx;                // column within LDS window (per lane)
      const unsigned short* arow0 =
          Wb + (((size_t)(b * 9 + tap) * COUT + co0 + m) * CIN);
      const unsigned short* arow1 = arow0 + (size_t)16 * CIN;
      const unsigned short* bcol  = &smem[(size_t)(s * 17 + cc) * CIN];
#pragma unroll 4
      for (int kc = 0; kc < CIN / 32; ++kc) {
        // A fragments: per lane two contiguous 16B global loads (K = kc*32 + hi*8 ..)
        union { u32x4 q[2]; v16bf v; } A0, A1, Bf;
        A0.q[0] = *reinterpret_cast<const u32x4*>(arow0 + kc * 32 + hi * 8);
        A0.q[1] = *reinterpret_cast<const u32x4*>(arow0 + kc * 32 + 16 + hi * 8);
        A1.q[0] = *reinterpret_cast<const u32x4*>(arow1 + kc * 32 + hi * 8);
        A1.q[1] = *reinterpret_cast<const u32x4*>(arow1 + kc * 32 + 16 + hi * 8);
        // B fragment: per lane two ds_load_b128 (K = kc*32 + hi*8 .. and kc*32 + 16 + hi*8 ..)
        const unsigned short* bp = bcol + kc * 32 + hi * 8;
        Bf.q[0] = *reinterpret_cast<const u32x4*>(bp);
        Bf.q[1] = *reinterpret_cast<const u32x4*>(bp + 16);
        acc0 = __builtin_amdgcn_wmma_f32_16x16x32_bf16(
            false, A0.v, false, Bf.v, (short)0, acc0, false, false);
        acc1 = __builtin_amdgcn_wmma_f32_16x16x32_bf16(
            false, A1.v, false, Bf.v, (short)0, acc1, false, false);
      }
    }
  }

  // store two 16x16 tiles: lane holds (co = co0 + r + hi*8 [+16], pixel n = lane&15)
  const int n    = lane & 15;
  const int xs   = 16 * xt + n;
  const int xmax = ex ? 32 : 33;
  if (xs < xmax) {
    const int x = ex + 2 * xs;
    float* obase = o65 + (((size_t)b * COUT + co0 + hi * 8) * H2 + y) * H2 + x;
#pragma unroll
    for (int r = 0; r < 8; ++r)
      obase[(size_t)r * H2 * H2] = acc0[r];
    float* obase2 = obase + (size_t)16 * H2 * H2;
#pragma unroll
    for (int r = 0; r < 8; ++r)
      obase2[(size_t)r * H2 * H2] = acc1[r];
  }
}

// ---------------- blur 4x4 + noise + bias + leaky_relu * sqrt(2) ----------------
__global__ void blur_kernel(const float* __restrict__ o65, const float* __restrict__ noise,
                            const float* __restrict__ nw, const float* __restrict__ lb,
                            float* __restrict__ out) {
  long long idx = (long long)blockIdx.x * 256 + threadIdx.x;
  const long long total = (long long)B_ * COUT * HO * HO;
  if (idx >= total) return;
  int x  = (int)(idx & 63);
  int yy = (int)((idx >> 6) & 63);
  int co = (int)((idx >> 12) & 511);
  int b  = (int)(idx >> 21);

  const float* src = o65 + ((size_t)b * COUT + co) * H2 * H2;
  const float k4[4] = {1.f, 3.f, 3.f, 1.f};
  float acc = 0.f;
#pragma unroll
  for (int by = 0; by < 4; ++by) {
    int sy = yy - 1 + by;
    if (sy < 0 || sy >= H2) continue;
    const float* row = src + (size_t)sy * H2;
    float racc = 0.f;
#pragma unroll
    for (int bx = 0; bx < 4; ++bx) {
      int sx = x - 1 + bx;
      if (sx < 0 || sx >= H2) continue;
      racc += k4[bx] * row[sx];
    }
    acc += k4[by] * racc;
  }
  acc *= (1.f / 16.f);
  acc += nw[0] * noise[((size_t)b * HO + yy) * HO + x];
  acc += lb[co];
  float r = (acc > 0.f) ? acc : 0.2f * acc;
  out[idx] = r * 1.41421356237309515f;
}

// ---------------- host launch ----------------
extern "C" void kernel_launch(void* const* d_in, const int* in_sizes, int n_in,
                              void* d_out, int out_size, void* d_ws, size_t ws_size,
                              hipStream_t stream) {
  (void)in_sizes; (void)n_in; (void)out_size; (void)ws_size;
  const float* input = (const float*)d_in[0];
  const float* w     = (const float*)d_in[1];
  const float* noise = (const float*)d_in[2];
  const float* aw    = (const float*)d_in[3];
  const float* ab    = (const float*)d_in[4];
  const float* cw    = (const float*)d_in[5];
  const float* nw    = (const float*)d_in[6];   // scalar
  const float* lb    = (const float*)d_in[7];
  float* out = (float*)d_out;

  char* base = (char*)d_ws;
  size_t off = 0;
  auto alloc = [&](size_t bytes) -> char* {
    off = (off + 255) & ~(size_t)255;
    char* p = base + off;
    off += bytes;
    return p;
  };
  float*          style = (float*)alloc((size_t)B_ * CIN * 4);
  float*          wsq   = (float*)alloc((size_t)COUT * CIN * 4);
  float*          dmod  = (float*)alloc((size_t)B_ * COUT * 4);
  unsigned short* Wb    = (unsigned short*)alloc((size_t)B_ * 9 * COUT * CIN * 2);
  unsigned short* XbT   = (unsigned short*)alloc((size_t)B_ * HH * WW * CIN * 2);
  float*          o65   = (float*)alloc((size_t)B_ * COUT * H2 * H2 * 4);

  const float lin_scale = 1.0f / sqrtf((float)LL);
  const float scale     = 1.0f / sqrtf((float)(CIN * 9));
  const float scale2    = scale * scale;

  style_kernel<<<(B_ * CIN + 255) / 256, 256, 0, stream>>>(w, aw, ab, style, lin_scale);
  wsq_kernel<<<(COUT * CIN + 255) / 256, 256, 0, stream>>>(cw, wsq);
  demod_kernel<<<(B_ * COUT + 255) / 256, 256, 0, stream>>>(style, wsq, dmod, scale2);
  {
    long long total = (long long)B_ * 9 * COUT * CIN;
    packw_kernel<<<(unsigned)((total + 255) / 256), 256, 0, stream>>>(cw, style, dmod, Wb, scale);
  }
  transpose_kernel<<<B_ * HH, 256, 0, stream>>>(input, XbT);
  upconv_kernel<<<dim3(5, H2, B_ * 2), 256, 0, stream>>>(Wb, XbT, o65);
  {
    long long total = (long long)B_ * COUT * HO * HO;
    blur_kernel<<<(unsigned)((total + 255) / 256), 256, 0, stream>>>(o65, noise, nw, lb, out);
  }
}
